// PhiAttention_48232482734526
// MI455X (gfx1250) — matrix operand order repaired and checked by reading, not compiled, not run on hardware
//
#include <hip/hip_runtime.h>

typedef __attribute__((ext_vector_type(16))) __bf16 v16bf;
typedef __attribute__((ext_vector_type(8)))  __bf16 v8bf;
typedef __attribute__((ext_vector_type(8)))  float  v8f;

#define HIDDEN 2560
#define NHEADS 32
#define HEADD  80
#define HD_PAD 96
#define ROTD   32
#define BATCH  2
#define SEQ    2048
#define MTOT   (BATCH*SEQ)      /* 4096 */
#define QKV_N  (3*HIDDEN)       /* 7680 */

static __device__ __forceinline__ v16bf pack16(v8bf lo, v8bf hi) {
  return __builtin_shufflevector(lo, hi, 0,1,2,3,4,5,6,7,8,9,10,11,12,13,14,15);
}

// ---------------------------------------------------------------- cast fp32 -> bf16
__global__ void phi_cast_bf16(const float* __restrict__ in, __bf16* __restrict__ out, int n) {
  int i = blockIdx.x * blockDim.x + threadIdx.x;
  if (i < n) out[i] = (__bf16)in[i];
}

// ------------------------------------------- transpose + cast: W[K][N] -> Wt[N][K] bf16
__global__ void phi_transpose_cast(const float* __restrict__ W, __bf16* __restrict__ Wt,
                                   int K, int N) {
  __shared__ float tile[32][33];
  const int nb = blockIdx.x * 32, kb = blockIdx.y * 32;
  const int tx = threadIdx.x, ty = threadIdx.y;   // block (32,8)
#pragma unroll
  for (int i = 0; i < 32; i += 8)
    tile[ty + i][tx] = W[(size_t)(kb + ty + i) * N + nb + tx];
  __syncthreads();
#pragma unroll
  for (int i = 0; i < 32; i += 8)
    Wt[(size_t)(nb + ty + i) * K + kb + tx] = (__bf16)tile[tx][ty + i];
}

// ---------------------------------------------------------------- WMMA GEMM
// C[M,N] f32 = A[M,K] (bf16, row-major) @ Bt[N,K] (bf16, B transposed) + bias[N]
// Block: 256 threads = 8 waves (2 x 4), wave tile 64x64, block tile 128x256, K-step 32.
__global__ __launch_bounds__(256) void phi_gemm_bf16_wmma(
    const __bf16* __restrict__ A, const __bf16* __restrict__ Bt,
    const float* __restrict__ bias, float* __restrict__ C,
    int M, int N, int K) {
  const int lane = threadIdx.x & 31;
  const int wave = threadIdx.x >> 5;
  const int l15 = lane & 15, g = lane >> 4;
  const int wm = wave & 1, wn = wave >> 1;
  const int row0 = blockIdx.y * 128 + wm * 64;
  const int col0 = blockIdx.x * 256 + wn * 64;

  v8f acc[4][4];
#pragma unroll
  for (int mi = 0; mi < 4; ++mi)
#pragma unroll
    for (int ni = 0; ni < 4; ++ni) acc[mi][ni] = (v8f){};

  for (int k0 = 0; k0 < K; k0 += 32) {
    v16bf a[4], b[4];
#pragma unroll
    for (int mi = 0; mi < 4; ++mi) {
      const __bf16* p = A + (size_t)(row0 + mi * 16 + l15) * K + k0 + g * 16;
      v8bf lo = *(const v8bf*)p;
      v8bf hi = *(const v8bf*)(p + 8);
      a[mi] = pack16(lo, hi);
    }
#pragma unroll
    for (int ni = 0; ni < 4; ++ni) {
      const __bf16* p = Bt + (size_t)(col0 + ni * 16 + l15) * K + k0 + g * 16;
      b[ni] = *(const v16bf*)p;
    }
#pragma unroll
    for (int mi = 0; mi < 4; ++mi)
#pragma unroll
      for (int ni = 0; ni < 4; ++ni)
        acc[mi][ni] = __builtin_amdgcn_wmma_f32_16x16x32_bf16(
            false, a[mi], false, b[ni], (short)0, acc[mi][ni], false, false);
  }

#pragma unroll
  for (int ni = 0; ni < 4; ++ni) {
    const int col = col0 + ni * 16 + l15;
    const float bv = bias[col];
#pragma unroll
    for (int mi = 0; mi < 4; ++mi) {
      const int rowb = row0 + mi * 16 + 8 * g;   // reg r -> row = rowb + r
#pragma unroll
      for (int r = 0; r < 8; ++r)
        C[(size_t)(rowb + r) * N + col] = acc[mi][ni][r] + bv;
    }
  }
}

// ------------------------------------- QKV scatter + RoPE + scale + V-transpose (bf16)
// fused[b][s][h][t][d] fp32 -> Q/K: [bh][s][96] (pad 80..95 = 0), Vt: [bh][80][s]
__global__ void phi_qkv_rope_scatter(const float* __restrict__ fused,
                                     const int* __restrict__ pos,
                                     __bf16* __restrict__ Q, __bf16* __restrict__ Kb,
                                     __bf16* __restrict__ Vt) {
  const int tid = blockIdx.x * 256 + threadIdx.x;
  const int TOTAL = BATCH * NHEADS * SEQ * 272;   // 96 q + 96 k + 80 v slots
  if (tid >= TOTAL) return;
  const int slot = tid % 272;
  const int gid = tid / 272;            // (b*NHEADS + h)*SEQ + s
  const int s = gid % SEQ;
  const int bh = gid / SEQ;
  const int h = bh & 31, b = bh >> 5;
  const size_t fbase = ((size_t)(b * SEQ + s)) * QKV_N + h * 240;

  if (slot < 192) {
    const int t = slot / 96;            // 0=q, 1=k
    const int d = slot % 96;
    float v = 0.f;
    if (d < HEADD) {
      v = fused[fbase + t * 80 + d];
      if (d < ROTD) {
        const int fi = d & 15;
        const float invf = __expf(-(float)fi * (9.210340372f / 16.0f)); // theta^(-2fi/32)
        const float ang = (float)pos[s] * invf;
        float sn, cs;
        __sincosf(ang, &sn, &cs);
        const float vp = fused[fbase + t * 80 + (d < 16 ? d + 16 : d - 16)];
        v = (d < 16) ? (v * cs - vp * sn) : (v * cs + vp * sn);
      }
      if (t == 0) v *= 0.1118033989f;   // 1/sqrt(80) folded into Q
    }
    __bf16* dst = (t == 0) ? Q : Kb;
    dst[((size_t)bh * SEQ + s) * HD_PAD + d] = (__bf16)v;
  } else {
    const int d = slot - 192;           // 0..79
    const float v = fused[fbase + 160 + d];
    Vt[((size_t)bh * HEADD + d) * SEQ + s] = (__bf16)v;
  }
}

// ---------------------------------------------------------------- flash attention
// Computes S^T = K·Q^T so P lands directly in the A-fragment layout for P·V.
// Block = 8 waves; each wave owns 16 q rows; 32 keys per iteration.
__global__ __launch_bounds__(256) void phi_flash_attn_wmma(
    const __bf16* __restrict__ Q, const __bf16* __restrict__ Kb,
    const __bf16* __restrict__ Vt, __bf16* __restrict__ Aout) {
  const int lane = threadIdx.x & 31;
  const int wave = threadIdx.x >> 5;
  const int l15 = lane & 15, g = lane >> 4;
  const int bh = blockIdx.y;
  const int b = bh >> 5, h = bh & 31;
  const int qbase = blockIdx.x * 128 + wave * 16;
  const int qrow = qbase + l15;          // this lane's q row (S^T column)

  // Q as B-fragments: lane = q row, 16 contiguous dims per chunk
  const __bf16* Qr = Q + ((size_t)bh * SEQ + qrow) * HD_PAD;
  v16bf bQ[3];
#pragma unroll
  for (int c = 0; c < 3; ++c) bQ[c] = *(const v16bf*)(Qr + c * 32 + g * 16);

  float mstate = -1e30f, lstate = 0.f;
  v8f o[5];
#pragma unroll
  for (int c5 = 0; c5 < 5; ++c5) o[c5] = (v8f){};

  const int jmax = (qbase + 15) >> 5;
  for (int j = 0; j <= jmax; ++j) {
    const int kb0 = j * 32;
    // K as A-fragments (2 key-halves x 3 d-chunks)
    v16bf aK[2][3];
#pragma unroll
    for (int f = 0; f < 2; ++f) {
      const __bf16* Krow = Kb + ((size_t)bh * SEQ + kb0 + f * 16 + l15) * HD_PAD;
#pragma unroll
      for (int c = 0; c < 3; ++c) {
        v8bf lo = *(const v8bf*)(Krow + c * 32 + g * 16);
        v8bf hi = *(const v8bf*)(Krow + c * 32 + g * 16 + 8);
        aK[f][c] = pack16(lo, hi);
      }
    }
    // S^T: D[key][qrow]; lane holds qrow=l15, reg r holds key f*16 + r + 8g
    v8f st0 = (v8f){}, st1 = (v8f){};
#pragma unroll
    for (int c = 0; c < 3; ++c) {
      st0 = __builtin_amdgcn_wmma_f32_16x16x32_bf16(false, aK[0][c], false, bQ[c],
                                                    (short)0, st0, false, false);
      st1 = __builtin_amdgcn_wmma_f32_16x16x32_bf16(false, aK[1][c], false, bQ[c],
                                                    (short)0, st1, false, false);
    }
    if (kb0 + 31 > qbase) {              // causal mask, only on the diagonal block
#pragma unroll
      for (int r = 0; r < 8; ++r) {
        const int key0 = kb0 + r + 8 * g;
        if (key0 > qrow)      st0[r] = -1e30f;
        if (key0 + 16 > qrow) st1[r] = -1e30f;
      }
    }
    // online softmax (row = qrow = l15; reduce over regs + the two g-groups)
    float mx = -1e30f;
#pragma unroll
    for (int r = 0; r < 8; ++r) mx = fmaxf(mx, fmaxf(st0[r], st1[r]));
    mx = fmaxf(mx, __shfl_xor(mx, 16, 32));
    const float mnew = fmaxf(mstate, mx);
    const float alpha = __expf(mstate - mnew);
    float p0[8], p1[8], lsum = 0.f;
#pragma unroll
    for (int r = 0; r < 8; ++r) {
      p0[r] = __expf(st0[r] - mnew);
      p1[r] = __expf(st1[r] - mnew);
      lsum += p0[r] + p1[r];
    }
    lsum += __shfl_xor(lsum, 16, 32);
    lstate = lstate * alpha + lsum;
    mstate = mnew;
    // P as A-fragment: element e -> key (e>>3)*16 + g*8 + (e&7) = st[e>>3] reg (e&7)
    v16bf ap;
#pragma unroll
    for (int e = 0; e < 8; ++e) { ap[e] = (__bf16)p0[e]; ap[e + 8] = (__bf16)p1[e]; }
    // per-output-row rescale factor: O reg r is qrow r+8g; factor lives at lane r+8g
    float ar[8];
#pragma unroll
    for (int r = 0; r < 8; ++r) ar[r] = __shfl(alpha, r + 8 * g, 32);
#pragma unroll
    for (int c5 = 0; c5 < 5; ++c5) {
      const __bf16* vp = Vt + ((size_t)bh * HEADD + c5 * 16 + l15) * SEQ + kb0 + g * 8;
      const v16bf bV = pack16(*(const v8bf*)vp, *(const v8bf*)(vp + 16));
#pragma unroll
      for (int r = 0; r < 8; ++r) o[c5][r] *= ar[r];
      o[c5] = __builtin_amdgcn_wmma_f32_16x16x32_bf16(false, ap, false, bV,
                                                      (short)0, o[c5], false, false);
    }
  }
  const float inv = 1.0f / lstate;
  float lr[8];
#pragma unroll
  for (int r = 0; r < 8; ++r) lr[r] = __shfl(inv, r + 8 * g, 32);
#pragma unroll
  for (int c5 = 0; c5 < 5; ++c5)
#pragma unroll
    for (int r = 0; r < 8; ++r) {
      const int srow = qbase + r + 8 * g;
      Aout[((size_t)(b * SEQ + srow)) * HIDDEN + h * HEADD + c5 * 16 + l15] =
          (__bf16)(o[c5][r] * lr[r]);
    }
}

// ---------------------------------------------------------------- host launch
extern "C" void kernel_launch(void* const* d_in, const int* in_sizes, int n_in,
                              void* d_out, int out_size, void* d_ws, size_t ws_size,
                              hipStream_t stream) {
  const float* hidden = (const float*)d_in[0];
  const int*   pos    = (const int*)d_in[1];
  const float* Wqkv   = (const float*)d_in[2];
  const float* bqkv   = (const float*)d_in[3];
  const float* Wd     = (const float*)d_in[4];
  const float* bd     = (const float*)d_in[5];
  float* out = (float*)d_out;

  char* ws = (char*)d_ws;
  size_t off = 0;
  float*  fused = (float*)(ws + off);  off += (size_t)MTOT * QKV_N * 4;        // 125.8 MB
  __bf16* Abf   = (__bf16*)fused;      // reuse fused region after scatter
  __bf16* Xbf   = (__bf16*)(ws + off); off += (size_t)MTOT * HIDDEN * 2;       // 21.0 MB
  __bf16* WqkvT = (__bf16*)(ws + off); off += (size_t)QKV_N * HIDDEN * 2;      // 39.3 MB
  __bf16* WdT   = (__bf16*)(ws + off); off += (size_t)HIDDEN * HIDDEN * 2;     // 13.1 MB
  __bf16* Qb    = (__bf16*)(ws + off); off += (size_t)BATCH * NHEADS * SEQ * HD_PAD * 2;
  __bf16* Kbf   = (__bf16*)(ws + off); off += (size_t)BATCH * NHEADS * SEQ * HD_PAD * 2;
  __bf16* Vt    = (__bf16*)(ws + off); off += (size_t)BATCH * NHEADS * HEADD * SEQ * 2;
  (void)ws_size; (void)in_sizes; (void)n_in; (void)out_size;

  // 1) casts / transposed weight casts
  phi_cast_bf16<<<(MTOT * HIDDEN + 255) / 256, 256, 0, stream>>>(hidden, Xbf, MTOT * HIDDEN);
  phi_transpose_cast<<<dim3(QKV_N / 32, HIDDEN / 32), dim3(32, 8), 0, stream>>>(
      Wqkv, WqkvT, HIDDEN, QKV_N);
  phi_transpose_cast<<<dim3(HIDDEN / 32, HIDDEN / 32), dim3(32, 8), 0, stream>>>(
      Wd, WdT, HIDDEN, HIDDEN);
  // 2) QKV projection
  phi_gemm_bf16_wmma<<<dim3(QKV_N / 256, MTOT / 128), 256, 0, stream>>>(
      Xbf, WqkvT, bqkv, fused, MTOT, QKV_N, HIDDEN);
  // 3) RoPE + scale + scatter to Q/K (padded) and V^T
  {
    const int total = BATCH * NHEADS * SEQ * 272;
    phi_qkv_rope_scatter<<<(total + 255) / 256, 256, 0, stream>>>(fused, pos, Qb, Kbf, Vt);
  }
  // 4) flash attention -> bf16 attn in [B*S][HIDDEN] (aliases fused)
  phi_flash_attn_wmma<<<dim3(SEQ / 128, BATCH * NHEADS), 256, 0, stream>>>(Qb, Kbf, Vt, Abf);
  // 5) output projection -> fp32 d_out
  phi_gemm_bf16_wmma<<<dim3(HIDDEN / 256, MTOT / 128), 256, 0, stream>>>(
      Abf, WdT, bd, out, MTOT, HIDDEN, HIDDEN);
}
